// Decoder_54004918780713
// MI455X (gfx1250) — compile-verified, matrix operand order checked by reading, not run on hardware
//
#include <hip/hip_runtime.h>
#include <stdint.h>

// ---------------- model constants ----------------
#define BB   8
#define SS   512          // S_SRC == S_TGT
#define DD   512
#define HH   8
#define DHH  64
#define FF   2048
#define LL   6
#define VV   32000
#define MM   (BB * SS)    // 4096 rows of activation
#define SCALE_F 0.125f    // 1/sqrt(64)

typedef __attribute__((ext_vector_type(16))) __bf16 v16bf;
typedef __attribute__((ext_vector_type(8)))  __bf16 v8bf;
typedef __attribute__((ext_vector_type(8)))  float  v8f;

__device__ __forceinline__ v8f wmma_bf16(v16bf a, v16bf b, v8f c) {
    // D = A(16x32 bf16) * B(32x16 bf16) + C(16x16 f32)
    return __builtin_amdgcn_wmma_f32_16x16x32_bf16(
        /*neg_a=*/false, a, /*neg_b=*/false, b,
        /*c_mod=*/(short)0, c, /*reuse_a=*/false, /*reuse_b=*/false);
}

// Assemble a 16-element fragment from two contiguous 16-byte chunks.
// Half-slot h holds K = ((h&8)<<1) + (h&7) + 8*hi -> chunk0 at K base hi*8,
// chunk1 at K base 16+hi*8 (8 contiguous elements each).
__device__ __forceinline__ v16bf load_frag(const __bf16* p0, const __bf16* p1) {
    v8bf lo = *(const v8bf*)p0;
    v8bf hi = *(const v8bf*)p1;
    return __builtin_shufflevector(lo, hi, 0, 1, 2, 3, 4, 5, 6, 7,
                                   8, 9, 10, 11, 12, 13, 14, 15);
}

// ---------------- converters ----------------
__global__ void cvt_f32_bf16_kernel(const float* __restrict__ in,
                                    __bf16* __restrict__ out, size_t n) {
    size_t i = (size_t)blockIdx.x * blockDim.x + threadIdx.x;
    size_t stride = (size_t)gridDim.x * blockDim.x;
    for (; i < n; i += stride) out[i] = (__bf16)in[i];
}

// batched weight transpose-convert: in [L][K][N] f32 -> out [L][N][K] bf16
__global__ void transpose_cvt_w_kernel(const float* __restrict__ in,
                                       __bf16* __restrict__ out,
                                       int Kd, int Nd, int layers) {
    size_t plane = (size_t)Kd * Nd;
    size_t total = plane * layers;
    size_t i = (size_t)blockIdx.x * blockDim.x + threadIdx.x;
    size_t stride = (size_t)gridDim.x * blockDim.x;
    for (; i < total; i += stride) {
        size_t l = i / plane, r = i % plane;
        size_t k = r / Nd, n = r % Nd;
        out[l * plane + n * (size_t)Kd + k] = (__bf16)in[i];
    }
}

// x = emb[target]; also bf16 mirror. grid = MM blocks, 128 threads
__global__ void embed_kernel(const int* __restrict__ target,
                             const float* __restrict__ emb,
                             float* __restrict__ x, __bf16* __restrict__ xbf) {
    int row = blockIdx.x;
    int tok = target[row];
    int t = threadIdx.x;
    const float* src = emb + (size_t)tok * DD;
#pragma unroll
    for (int i = 0; i < 4; ++i) {
        int c = t + 128 * i;
        float v = src[c];
        x[(size_t)row * DD + c]   = v;
        xbf[(size_t)row * DD + c] = (__bf16)v;
    }
}

// ---------------- WMMA GEMM: Y[M,N] = X[M,K] @ W + bias ----------------
// W is pre-transposed: WT[N][K] bf16 row-major.
// block = 128 threads (4 waves), tile 128(M) x 64(N), K-step 32,
// compile-time N/K, 2-stage LDS double buffering, sched_barrier-pinned
// load / compute / store phases. Per thread per step: 4+2 global_load_b128,
// 4+2 ds_store_b128. Wave w owns rows [32w,32w+32): 8 WMMAs per K-step.
enum { EPI_F32 = 0, EPI_F32_RELU = 1, EPI_BF16 = 2, EPI_BF16_RELU = 3, EPI_BF16_VT = 4 };

template <int EPI, bool HAS_BIAS, int N, int K>
__global__ __launch_bounds__(128) void gemm_wmma_kernel(
        const __bf16* __restrict__ X, const __bf16* __restrict__ WT,
        const float* __restrict__ bias, void* __restrict__ Y) {
    __shared__ __bf16 As[2][128 * 32];   // [m][k] row-major
    __shared__ __bf16 Bs[2][64 * 32];    // [n][k] n-major
    const int n0 = blockIdx.x * 64;
    const int m0 = blockIdx.y * 128;
    const int t = threadIdx.x;           // 0..127
    const int lane = t & 31;
    const int w = t >> 5;                // wave id 0..3
    const int m = lane & 15;
    const int hi = lane >> 4;

    v8f acc[2][4];
#pragma unroll
    for (int mi = 0; mi < 2; ++mi)
#pragma unroll
        for (int j = 0; j < 4; ++j)
#pragma unroll
            for (int e = 0; e < 8; ++e) acc[mi][j][e] = 0.0f;

    v8bf xv[4], wv[2];
    const __bf16* xp = X  + (size_t)(m0 + t) * K;                     // own X row
    const __bf16* wp = WT + (size_t)(n0 + (t >> 1)) * K + (t & 1) * 16; // own WT half-row

    auto load_tile = [&]() {
#pragma unroll
        for (int i = 0; i < 4; ++i) xv[i] = ((const v8bf*)xp)[i];
        wv[0] = ((const v8bf*)wp)[0];
        wv[1] = ((const v8bf*)wp)[1];
    };
    auto store_tile = [&](int buf) {
        v8bf* ar = (v8bf*)&As[buf][t * 32];
#pragma unroll
        for (int i = 0; i < 4; ++i) ar[i] = xv[i];
        v8bf* br = (v8bf*)&Bs[buf][(t >> 1) * 32 + (t & 1) * 16];
        br[0] = wv[0];
        br[1] = wv[1];
    };
    auto compute = [&](int buf) {
        v16bf a[2];
#pragma unroll
        for (int mi = 0; mi < 2; ++mi) {
            const __bf16* base = &As[buf][(32 * w + 16 * mi + m) * 32 + hi * 8];
            a[mi] = load_frag(base, base + 16);
        }
#pragma unroll
        for (int j = 0; j < 4; ++j) {
            const __bf16* bb = &Bs[buf][(j * 16 + m) * 32 + hi * 8];
            v16bf b = load_frag(bb, bb + 16);
            acc[0][j] = wmma_bf16(a[0], b, acc[0][j]);
            acc[1][j] = wmma_bf16(a[1], b, acc[1][j]);
        }
    };

    load_tile();
    store_tile(0);
    __syncthreads();
    int cur = 0;
#pragma unroll 2
    for (int k0 = 32; k0 < K; k0 += 32) {
        xp += 32;
        wp += 32;
        load_tile();                        // issue global loads (in-flight)
        __builtin_amdgcn_sched_barrier(0);  // keep loads above the WMMAs
        compute(cur);                       // 8 WMMAs overlap the loads
        __builtin_amdgcn_sched_barrier(0);  // keep stores below the WMMAs
        store_tile(cur ^ 1);                // loadcnt waits land after compute
        __syncthreads();
        cur ^= 1;
    }
    compute(cur);

    // epilogue: bias hoisted to 4 per-lane registers (column-invariant)
    float bj[4];
#pragma unroll
    for (int j = 0; j < 4; ++j)
        bj[j] = HAS_BIAS ? bias[n0 + j * 16 + m] : 0.0f;

#pragma unroll
    for (int mi = 0; mi < 2; ++mi)
#pragma unroll
        for (int j = 0; j < 4; ++j)
#pragma unroll
            for (int i = 0; i < 8; ++i) {
                int row = m0 + 32 * w + 16 * mi + i + 8 * hi;
                int col = n0 + j * 16 + m;
                float v = acc[mi][j][i] + bj[j];
                if (EPI == EPI_F32_RELU || EPI == EPI_BF16_RELU) v = fmaxf(v, 0.0f);
                if (EPI == EPI_BF16 || EPI == EPI_BF16_RELU) {
                    ((__bf16*)Y)[(size_t)row * N + col] = (__bf16)v;
                } else if (EPI == EPI_BF16_VT) {
                    // V projection: write Vt[b][h][dh][s]  (N == DD here)
                    int b_ = row >> 9, s_ = row & (SS - 1);
                    int h_ = col >> 6, d_ = col & (DHH - 1);
                    ((__bf16*)Y)[(((size_t)(b_ * HH + h_)) * DHH + d_) * SS + s_] = (__bf16)v;
                } else {
                    ((float*)Y)[(size_t)row * N + col] = v;
                }
            }
}

// ---------------- fused attention ----------------
// grid (SS/16, H, B), block = 32 (one wave).
// Q,K: [B, S, D] bf16 (head h at cols [64h,64h+64)); V: Vt[b][h][dh][s] bf16.
template <bool CAUSAL>
__global__ __launch_bounds__(32) void attn_kernel(
        const __bf16* __restrict__ Q, const __bf16* __restrict__ K,
        const __bf16* __restrict__ Vt, __bf16* __restrict__ O) {
    __shared__ float  sc[16][SS];   // 32 KB scores / exp values
    __shared__ __bf16 pb[16][SS];   // 16 KB normalized probs (bf16)
    const int qt = blockIdx.x, h = blockIdx.y, b = blockIdx.z;
    const int lane = threadIdx.x;
    const int m = lane & 15;
    const int hi = lane >> 4;

    const __bf16* Qb  = Q  + ((size_t)b * SS) * DD + h * DHH;
    const __bf16* Kb  = K  + ((size_t)b * SS) * DD + h * DHH;
    const __bf16* Vtb = Vt + ((size_t)(b * HH + h)) * DHH * SS;

    // A fragments of Q tile [16 x 64] -> two 16x32 frags (vector loads)
    v16bf aq[2];
#pragma unroll
    for (int kk = 0; kk < 2; ++kk) {
        const __bf16* base = Qb + (size_t)(qt * 16 + m) * DD + kk * 32 + hi * 8;
        aq[kk] = load_frag(base, base + 16);
    }

    // scores = Q @ K^T * scale  (loop over 16-wide key tiles)
    for (int nt = 0; nt < SS / 16; ++nt) {
        if (nt + 1 < SS / 16)
            __builtin_prefetch(Kb + (size_t)((nt + 1) * 16 + m) * DD, 0, 0);
        v8f accs;
#pragma unroll
        for (int e = 0; e < 8; ++e) accs[e] = 0.0f;
#pragma unroll
        for (int kk = 0; kk < 2; ++kk) {
            // B[k][n] = K[nt*16+n][k]; per-lane n fixed -> contiguous k chunks
            const __bf16* base = Kb + (size_t)(nt * 16 + m) * DD + kk * 32 + hi * 8;
            v16bf bk = load_frag(base, base + 16);
            accs = wmma_bf16(aq[kk], bk, accs);
        }
#pragma unroll
        for (int i = 0; i < 8; ++i) {
            int row = i + 8 * hi;
            int col = nt * 16 + m;
            float s = accs[i] * SCALE_F;
            if (CAUSAL && col > qt * 16 + row) s = -INFINITY;
            sc[row][col] = s;
        }
    }
    __syncthreads();

    // softmax per row; lanes (m, hi) split each row in halves, combine via shfl
    {
        float mx = -INFINITY;
        for (int c = hi * (SS / 2); c < (hi + 1) * (SS / 2); ++c)
            mx = fmaxf(mx, sc[m][c]);
        mx = fmaxf(mx, __shfl_xor(mx, 16));
        float sum = 0.0f;
        for (int c = hi * (SS / 2); c < (hi + 1) * (SS / 2); ++c) {
            float e = __expf(sc[m][c] - mx);
            sc[m][c] = e;
            sum += e;
        }
        sum += __shfl_xor(sum, 16);
        float inv = 1.0f / sum;
        for (int c = hi * (SS / 2); c < (hi + 1) * (SS / 2); ++c)
            pb[m][c] = (__bf16)(sc[m][c] * inv);
    }
    __syncthreads();

    // O = P @ V : M=16, N=64 (4 subtiles), K=SS in steps of 32
    v8f o[4];
#pragma unroll
    for (int j = 0; j < 4; ++j)
#pragma unroll
        for (int e = 0; e < 8; ++e) o[j][e] = 0.0f;

    for (int kb = 0; kb < SS / 32; ++kb) {
        const __bf16* pa = &pb[m][kb * 32 + hi * 8];
        v16bf ap = load_frag(pa, pa + 16);
#pragma unroll
        for (int j = 0; j < 4; ++j) {
            // B[k][n] = V[kb*32+k][j*16+n] = Vt[j*16+n][kb*32+k] : contiguous k
            const __bf16* base = Vtb + (size_t)(j * 16 + m) * SS + kb * 32 + hi * 8;
            v16bf bv = load_frag(base, base + 16);
            o[j] = wmma_bf16(ap, bv, o[j]);
        }
    }

    __bf16* Ob = O + ((size_t)b * SS + qt * 16) * DD + h * DHH;
#pragma unroll
    for (int j = 0; j < 4; ++j)
#pragma unroll
        for (int i = 0; i < 8; ++i) {
            int row = i + 8 * hi;
            int col = j * 16 + m;
            Ob[(size_t)row * DD + col] = (__bf16)o[j][i];
        }
}

// ---------------- residual add + LayerNorm ----------------
__global__ void add_ln_kernel(const float* __restrict__ res,
                              const float* __restrict__ delta,
                              const float* __restrict__ g,
                              const float* __restrict__ beta,
                              float* __restrict__ xo, __bf16* __restrict__ xbf) {
    __shared__ float red[128];
    const int row = blockIdx.x;
    const int t = threadIdx.x;
    const float* r = res + (size_t)row * DD;
    const float* dl = delta + (size_t)row * DD;

    float v[4];
    float s = 0.0f;
#pragma unroll
    for (int i = 0; i < 4; ++i) {
        int c = t + 128 * i;
        v[i] = r[c] + dl[c];
        s += v[i];
    }
    red[t] = s;
    __syncthreads();
    for (int off = 64; off > 0; off >>= 1) {
        if (t < off) red[t] += red[t + off];
        __syncthreads();
    }
    float mu = red[0] * (1.0f / DD);
    __syncthreads();

    float s2 = 0.0f;
#pragma unroll
    for (int i = 0; i < 4; ++i) {
        float d = v[i] - mu;
        s2 += d * d;
    }
    red[t] = s2;
    __syncthreads();
    for (int off = 64; off > 0; off >>= 1) {
        if (t < off) red[t] += red[t + off];
        __syncthreads();
    }
    float var = red[0] * (1.0f / DD);
    float inv = rsqrtf(var + 1e-5f);
#pragma unroll
    for (int i = 0; i < 4; ++i) {
        int c = t + 128 * i;
        float y = (v[i] - mu) * inv * g[c] + beta[c];
        xo[(size_t)row * DD + c] = y;
        xbf[(size_t)row * DD + c] = (__bf16)y;
    }
}

// ---------------- host side ----------------
extern "C" void kernel_launch(void* const* d_in, const int* in_sizes, int n_in,
                              void* d_out, int out_size, void* d_ws, size_t ws_size,
                              hipStream_t stream) {
    const float* encoded = (const float*)d_in[0];
    const int*   target  = (const int*)d_in[1];
    const float* emb     = (const float*)d_in[2];
    const float* Wq_m = (const float*)d_in[3];  const float* bq_m = (const float*)d_in[4];
    const float* Wk_m = (const float*)d_in[5];  const float* bk_m = (const float*)d_in[6];
    const float* Wv_m = (const float*)d_in[7];  const float* bv_m = (const float*)d_in[8];
    const float* Wo_m = (const float*)d_in[9];  const float* bo_m = (const float*)d_in[10];
    const float* Wq_s = (const float*)d_in[11]; const float* bq_s = (const float*)d_in[12];
    const float* Wk_s = (const float*)d_in[13]; const float* bk_s = (const float*)d_in[14];
    const float* Wv_s = (const float*)d_in[15]; const float* bv_s = (const float*)d_in[16];
    const float* Wo_s = (const float*)d_in[17]; const float* bo_s = (const float*)d_in[18];
    const float* W1   = (const float*)d_in[19]; const float* b1   = (const float*)d_in[20];
    const float* W2   = (const float*)d_in[21]; const float* b2   = (const float*)d_in[22];
    const float* ln_g = (const float*)d_in[23]; const float* ln_b = (const float*)d_in[24];

    // -------- workspace carve-up (256B aligned) --------
    size_t off = 0;
    auto carve = [&](size_t bytes) -> char* {
        char* p = (char*)d_ws + off;
        off += (bytes + 255) & ~(size_t)255;
        return p;
    };
    const size_t nDD = (size_t)LL * DD * DD;
    const size_t nDF = (size_t)LL * DD * FF;

    // all weights stored transposed: WT[N][K] bf16
    __bf16* WqT_m = (__bf16*)carve(nDD * 2);
    __bf16* WkT_m = (__bf16*)carve(nDD * 2);
    __bf16* WvT_m = (__bf16*)carve(nDD * 2);
    __bf16* WoT_m = (__bf16*)carve(nDD * 2);
    __bf16* WqT_s = (__bf16*)carve(nDD * 2);
    __bf16* WkT_s = (__bf16*)carve(nDD * 2);
    __bf16* WvT_s = (__bf16*)carve(nDD * 2);
    __bf16* WoT_s = (__bf16*)carve(nDD * 2);
    __bf16* W1T   = (__bf16*)carve(nDF * 2);     // [F][D]
    __bf16* W2T   = (__bf16*)carve(nDF * 2);     // [D][F]
    __bf16* emb_bf = (__bf16*)carve((size_t)VV * DD * 2);  // [V][D] == unembed WT
    float*  x_f    = (float*) carve((size_t)MM * DD * 4);
    __bf16* x_bf   = (__bf16*)carve((size_t)MM * DD * 2);
    __bf16* enc_bf = (__bf16*)carve((size_t)MM * DD * 2);
    __bf16* q_bf   = (__bf16*)carve((size_t)MM * DD * 2);
    __bf16* k_bf   = (__bf16*)carve((size_t)MM * DD * 2);
    __bf16* vt_bf  = (__bf16*)carve((size_t)MM * DD * 2);  // Vt[b][h][dh][s]
    __bf16* attn_bf = (__bf16*)carve((size_t)MM * DD * 2);
    float*  delta_f = (float*) carve((size_t)MM * DD * 4);
    __bf16* h_bf    = (__bf16*)carve((size_t)MM * FF * 2);
    (void)ws_size;

    // -------- one-time conversions (weights transposed to [N][K]) --------
    dim3 cb(256), cg(2048);
    transpose_cvt_w_kernel<<<cg, cb, 0, stream>>>(Wq_m, WqT_m, DD, DD, LL);
    transpose_cvt_w_kernel<<<cg, cb, 0, stream>>>(Wk_m, WkT_m, DD, DD, LL);
    transpose_cvt_w_kernel<<<cg, cb, 0, stream>>>(Wv_m, WvT_m, DD, DD, LL);
    transpose_cvt_w_kernel<<<cg, cb, 0, stream>>>(Wo_m, WoT_m, DD, DD, LL);
    transpose_cvt_w_kernel<<<cg, cb, 0, stream>>>(Wq_s, WqT_s, DD, DD, LL);
    transpose_cvt_w_kernel<<<cg, cb, 0, stream>>>(Wk_s, WkT_s, DD, DD, LL);
    transpose_cvt_w_kernel<<<cg, cb, 0, stream>>>(Wv_s, WvT_s, DD, DD, LL);
    transpose_cvt_w_kernel<<<cg, cb, 0, stream>>>(Wo_s, WoT_s, DD, DD, LL);
    transpose_cvt_w_kernel<<<cg, cb, 0, stream>>>(W1, W1T, DD, FF, LL);
    transpose_cvt_w_kernel<<<cg, cb, 0, stream>>>(W2, W2T, FF, DD, LL);
    cvt_f32_bf16_kernel<<<cg, cb, 0, stream>>>(emb, emb_bf, (size_t)VV * DD);
    cvt_f32_bf16_kernel<<<cg, cb, 0, stream>>>(encoded, enc_bf, (size_t)MM * DD);

    embed_kernel<<<MM, 128, 0, stream>>>(target, emb, x_f, x_bf);

    const dim3 gemmB(128);
    const dim3 gDD(DD / 64, MM / 128);   // N=512
    const dim3 gDF(FF / 64, MM / 128);   // N=2048
    const dim3 gDV(VV / 64, MM / 128);   // N=32000
    const dim3 attnG(SS / 16, HH, BB);

    for (int l = 0; l < LL; ++l) {
        const size_t wo = (size_t)l * DD * DD;
        const size_t fo = (size_t)l * DD * FF;
        // ---- memory attention ----
        gemm_wmma_kernel<EPI_BF16, true, DD, DD><<<gDD, gemmB, 0, stream>>>(x_bf,   WqT_m + wo, bq_m + l * DD, q_bf);
        gemm_wmma_kernel<EPI_BF16, true, DD, DD><<<gDD, gemmB, 0, stream>>>(enc_bf, WkT_m + wo, bk_m + l * DD, k_bf);
        gemm_wmma_kernel<EPI_BF16_VT, true, DD, DD><<<gDD, gemmB, 0, stream>>>(enc_bf, WvT_m + wo, bv_m + l * DD, vt_bf);
        attn_kernel<false><<<attnG, 32, 0, stream>>>(q_bf, k_bf, vt_bf, attn_bf);
        gemm_wmma_kernel<EPI_F32, true, DD, DD><<<gDD, gemmB, 0, stream>>>(attn_bf, WoT_m + wo, bo_m + l * DD, delta_f);
        add_ln_kernel<<<MM, 128, 0, stream>>>(x_f, delta_f, ln_g, ln_b, x_f, x_bf);
        // ---- causal self attention ----
        gemm_wmma_kernel<EPI_BF16, true, DD, DD><<<gDD, gemmB, 0, stream>>>(x_bf, WqT_s + wo, bq_s + l * DD, q_bf);
        gemm_wmma_kernel<EPI_BF16, true, DD, DD><<<gDD, gemmB, 0, stream>>>(x_bf, WkT_s + wo, bk_s + l * DD, k_bf);
        gemm_wmma_kernel<EPI_BF16_VT, true, DD, DD><<<gDD, gemmB, 0, stream>>>(x_bf, WvT_s + wo, bv_s + l * DD, vt_bf);
        attn_kernel<true><<<attnG, 32, 0, stream>>>(q_bf, k_bf, vt_bf, attn_bf);
        gemm_wmma_kernel<EPI_F32, true, DD, DD><<<gDD, gemmB, 0, stream>>>(attn_bf, WoT_s + wo, bo_s + l * DD, delta_f);
        add_ln_kernel<<<MM, 128, 0, stream>>>(x_f, delta_f, ln_g, ln_b, x_f, x_bf);
        // ---- FFN (double ReLU) ----
        gemm_wmma_kernel<EPI_BF16_RELU, true, FF, DD><<<gDF, gemmB, 0, stream>>>(x_bf, W1T + fo, b1 + l * FF, h_bf);
        gemm_wmma_kernel<EPI_F32_RELU, true, DD, FF><<<gDD, gemmB, 0, stream>>>(h_bf, W2T + fo, b2 + l * DD, delta_f);
        add_ln_kernel<<<MM, 128, 0, stream>>>(x_f, delta_f, ln_g, ln_b, x_f, x_bf);
    }

    // ---- unembed: logits = x @ emb^T -> d_out [MM, VV] f32 (WT == emb itself) ----
    gemm_wmma_kernel<EPI_F32, false, VV, DD><<<gDV, gemmB, 0, stream>>>(x_bf, emb_bf, nullptr, (float*)d_out);
}
